// LSTMModel_1717986918849
// MI455X (gfx1250) — matrix operand order repaired and checked by reading, not compiled, not run on hardware
//
#include <hip/hip_runtime.h>

#define B_BATCH 8192
#define T_SEQ   512
#define IN_DIM  5
#define H_DIM   32

typedef __attribute__((ext_vector_type(16))) __bf16 v16bf;
typedef __attribute__((ext_vector_type(8)))  __bf16 v8bf;
typedef __attribute__((ext_vector_type(8)))  float  v8f;

// ---- activations: gfx1250 V_TANH_F32 (single trans op), guarded ----
__device__ __forceinline__ float tanh_sw(float x) {
    return 2.0f * __builtin_amdgcn_rcpf(1.0f + __expf(-2.0f * x)) - 1.0f;
}
#if __has_builtin(__builtin_amdgcn_tanhf)
#define HW_TANH(x) __builtin_amdgcn_tanhf(x)
#elif __has_builtin(__builtin_amdgcn_tanh_f32)
#define HW_TANH(x) __builtin_amdgcn_tanh_f32(x)
#else
#define HW_TANH(x) tanh_sw(x)
#endif

// Sigmoid with the 0.5 input scale pre-folded into the weights:
// acc already holds x/2, so sigmoid(x) = 0.5 + 0.5*tanh(acc) -> 1 trans + 1 fma.
__device__ __forceinline__ v8f sigv_prescaled(v8f x) {
    v8f r;
#pragma unroll
    for (int i = 0; i < 8; ++i) r[i] = __builtin_fmaf(0.5f, HW_TANH(x[i]), 0.5f);
    return r;
}
__device__ __forceinline__ v8f tanhv(v8f x) {
    v8f r;
#pragma unroll
    for (int i = 0; i < 8; ++i) r[i] = HW_TANH(x[i]);
    return r;
}

// One wave (32 lanes) per block. Each wave owns 16 batch rows and the full
// 128-wide gate vector: 8 N-tiles, 2 chained bf16 WMMAs each per timestep.
__launch_bounds__(32, 1)
__global__ void lstm_wmma_kernel(const float* __restrict__ x,
                                 const float* __restrict__ W_ih,
                                 const float* __restrict__ W_hh,
                                 const float* __restrict__ b_ih,
                                 const float* __restrict__ b_hh,
                                 const float* __restrict__ W_fc,
                                 const float* __restrict__ b_fc,
                                 float* __restrict__ out)
{
    // per-wave transpose staging: h as [row 0..15][col 0..31] bf16
    __shared__ __align__(16) __bf16 hbuf[16 * 32];

    const int lane = threadIdx.x;   // wave32: 0..31
    const int lm   = lane & 15;
    const int half = lane >> 4;     // A frag: 0 -> K 0..15 ; 1 -> K 16..31
    const int rowBase = blockIdx.x * 16;

    // ---------- resident weights (B fragments) ----------
    // whh[n]: W_hh^T tile n: B[k][j] = W_hh[16n+j][k], elem e <-> k = e + 16*half
    // wih[n]: W_ih^T padded to K=32; row K=5 carries (b_ih+b_hh); rows 6..31 zero.
    // i/f/o gate tiles (n = 0..3, 6, 7) are pre-scaled by 0.5 for the sigmoid.
    v16bf whh[8];
    v16bf wih[8];
#pragma unroll
    for (int n = 0; n < 8; ++n) {
        const float ws = (n == 4 || n == 5) ? 1.0f : 0.5f;   // g-gate unscaled
        const int g = 16 * n + lm;                 // gate column of this lane
        const float* wr = W_hh + g * H_DIM + 16 * half;
#pragma unroll
        for (int v = 0; v < 8; ++v) {
            float2 w = *(const float2*)(wr + 2 * v);   // 8B aligned
            whh[n][2 * v]     = (__bf16)(w.x * ws);
            whh[n][2 * v + 1] = (__bf16)(w.y * ws);
        }
        v16bf wi;
#pragma unroll
        for (int e = 0; e < 16; ++e) wi[e] = (__bf16)0.0f;
        if (half == 0) {                           // only K rows 0..5 nonzero
#pragma unroll
            for (int k = 0; k < IN_DIM; ++k) wi[k] = (__bf16)(W_ih[g * IN_DIM + k] * ws);
            wi[IN_DIM] = (__bf16)((b_ih[g] + b_hh[g]) * ws);   // bias row (K=5)
        }
        wih[n] = wi;
    }

    // ---------- recurrent state ----------
    v16bf ha, xa;                                  // A fragments (bf16)
#pragma unroll
    for (int e = 0; e < 16; ++e) { ha[e] = (__bf16)0.0f; xa[e] = (__bf16)0.0f; }
    xa[IN_DIM] = (__bf16)1.0f;   // multiplies bias row (K=5; on upper half K=21, a zero B row)
    v8f c0 = {}, c1 = {};                          // cell, cols 0..15 / 16..31
    v8f h0 = {}, h1 = {};                          // hidden (C layout)
    const v8f vzero = {};

    // All 32 lanes load row lm's x (upper half's values hit zero B rows 16..20,
    // so they contribute nothing) -> no exec-mask churn in the loop.
    const float* xptr = x + (size_t)(rowBase + lm) * T_SEQ * IN_DIM;
    float xf[IN_DIM];                              // software-pipelined x stage
#pragma unroll
    for (int k = 0; k < IN_DIM; ++k) xf[k] = xptr[k];

#pragma unroll 2
    for (int t = 0; t < T_SEQ; ++t) {
        // consume x loaded during the previous iteration
#pragma unroll
        for (int k = 0; k < IN_DIM; ++k) xa[k] = (__bf16)xf[k];

        // issue loads for t+1 now; their wait lands a whole iteration later
        xptr += (t < T_SEQ - 1) ? IN_DIM : 0;
#pragma unroll
        for (int k = 0; k < IN_DIM; ++k) xf[k] = xptr[k];

        // gates = (x,1)*[W_ih^T;bias] + h*W_hh^T   (fp32 accum, 8 N-tiles)
        v8f acc[8];
#pragma unroll
        for (int n = 0; n < 8; ++n) {
            v8f tmp = __builtin_amdgcn_wmma_f32_16x16x32_bf16(
                false, xa, false, wih[n], (short)0, vzero, false, false);
            acc[n] = __builtin_amdgcn_wmma_f32_16x16x32_bf16(
                false, ha, false, whh[n], (short)0, tmp, false, false);
        }

        // gate order i,f,g,o along the 128 dim -> tiles (0,1)(2,3)(4,5)(6,7)
        v8f ig0 = sigv_prescaled(acc[0]), ig1 = sigv_prescaled(acc[1]);
        v8f fg0 = sigv_prescaled(acc[2]), fg1 = sigv_prescaled(acc[3]);
        v8f gg0 = tanhv(acc[4]),          gg1 = tanhv(acc[5]);
        v8f og0 = sigv_prescaled(acc[6]), og1 = sigv_prescaled(acc[7]);

        c0 = fg0 * c0 + ig0 * gg0;
        c1 = fg1 * c1 + ig1 * gg1;
        h0 = og0 * tanhv(c0);
        h1 = og1 * tanhv(c1);

        // transpose h: C layout (row in VGPR, col in lane) -> A layout via LDS.
        // DS ops are in-order within a wave; single wave per block -> no barrier.
#pragma unroll
        for (int r = 0; r < 8; ++r) {
            const int row = r + 8 * half;
            hbuf[row * 32 + lm]      = (__bf16)h0[r];
            hbuf[row * 32 + lm + 16] = (__bf16)h1[r];
        }
        v8bf lo = *(const v8bf*)&hbuf[lm * 32 + 16 * half];       // K 0..7 / 16..23
        v8bf hi = *(const v8bf*)&hbuf[lm * 32 + 16 * half + 8];   // K 8..15 / 24..31
        ha = __builtin_shufflevector(lo, hi, 0, 1, 2, 3, 4, 5, 6, 7,
                                             8, 9, 10, 11, 12, 13, 14, 15);
    }

    // ---------- head: out[b] = h[b,:] . W_fc[0,:] + b_fc ----------
    const float wlo = W_fc[lm];
    const float whi = W_fc[lm + 16];
    const float bfc = b_fc[0];
#pragma unroll
    for (int r = 0; r < 8; ++r) {
        float s = h0[r] * wlo + h1[r] * whi;     // partial over this lane's 2 cols
        s += __shfl_xor(s, 1, 32);               // reduce across 16-lane half
        s += __shfl_xor(s, 2, 32);
        s += __shfl_xor(s, 4, 32);
        s += __shfl_xor(s, 8, 32);
        if (lm == 0) out[rowBase + 8 * half + r] = s + bfc;
    }
}

extern "C" void kernel_launch(void* const* d_in, const int* in_sizes, int n_in,
                              void* d_out, int out_size, void* d_ws, size_t ws_size,
                              hipStream_t stream) {
    (void)in_sizes; (void)n_in; (void)d_ws; (void)ws_size; (void)out_size;
    const float* x    = (const float*)d_in[0];
    const float* W_ih = (const float*)d_in[1];
    const float* W_hh = (const float*)d_in[2];
    const float* b_ih = (const float*)d_in[3];
    const float* b_hh = (const float*)d_in[4];
    const float* W_fc = (const float*)d_in[5];
    const float* b_fc = (const float*)d_in[6];
    float* out = (float*)d_out;

    dim3 grid(B_BATCH / 16);   // 512 independent wave-tiles of 16 batch rows
    dim3 block(32);            // one wave32 per block
    lstm_wmma_kernel<<<grid, block, 0, stream>>>(x, W_ih, W_hh, b_ih, b_hh,
                                                 W_fc, b_fc, out);
}
